// CompNet_41695542509613
// MI455X (gfx1250) — matrix operand order repaired and checked by reading, not compiled, not run on hardware
//
#include <hip/hip_runtime.h>

typedef __attribute__((ext_vector_type(16))) __bf16 v16bf;
typedef __attribute__((ext_vector_type(8)))  float  v8f;

#define HDIM   64
#define DDIM   700
#define DPAD   704      // 22 * 32, zero-padded K
#define TSTEPS 250
#define BATCH  256
#define NCLS   20
#define T_SKIP 15
#define TAU_INV 0.5f

#define MBLK 128
#define KC   32
#define SPITCH 36       // bf16 elems; 72B row stride: 8B-aligned stores, conflict-free frags

// ---------------------------------------------------------------------------
// Kernel 0: W1eff = Wr[:, :64] @ W1  (bf16, K zero-padded to 704)
//           b_eff = Wr[:, :64] @ b1 + br
// ---------------------------------------------------------------------------
__global__ void prep_kernel(const float* __restrict__ W1, const float* __restrict__ b1,
                            const float* __restrict__ Wr, const float* __restrict__ br,
                            __bf16* __restrict__ W1eff, float* __restrict__ beff) {
  int id = blockIdx.x * blockDim.x + threadIdx.x;
  if (id < HDIM) {
    float s = br[id];
    for (int j = 0; j < HDIM; ++j) s += Wr[id * 2 * HDIM + j] * b1[j];
    beff[id] = s;
  }
  if (id >= HDIM * DPAD) return;
  int h = id / DPAD, k = id % DPAD;
  float s = 0.f;
  if (k < DDIM) {
    for (int j = 0; j < HDIM; ++j) s += Wr[h * 2 * HDIM + j] * W1[j * DDIM + k];
  }
  W1eff[h * DPAD + k] = (__bf16)s;
}

// ---------------------------------------------------------------------------
// Kernel 1: Z = bf16(X) @ W1eff^T + b_eff     (M=64000, N=64, K=704)
// One block = 128x64 tile, 8 waves; wave = one 16-row stripe x 4 N-tiles.
// Staging: branch-free global_load_b128 (clamped tail) -> cvt_pk_bf16 -> b64 LDS.
// ---------------------------------------------------------------------------
__global__ void __launch_bounds__(256)
gemm_kernel(const float* __restrict__ X, const __bf16* __restrict__ W1eff,
            const float* __restrict__ beff, float* __restrict__ Z) {
  __shared__ alignas(16) __bf16 Xs[MBLK * SPITCH];
  __shared__ alignas(16) __bf16 Ws[HDIM * SPITCH];
  const int tid  = threadIdx.x;
  const int wave = tid >> 5;
  const int lane = tid & 31;
  const int nl   = lane & 15;
  const int hl   = lane >> 4;          // lane half (wave32)
  const long row0 = (long)blockIdx.x * MBLK;

  v8f c[4];
  for (int nt = 0; nt < 4; ++nt) {
    float bv = beff[nt * 16 + nl];     // fold bias into accumulator init
    for (int r = 0; r < 8; ++r) c[nt][r] = bv;
  }

  for (int kc = 0; kc < DPAD / KC; ++kc) {
    const int k0 = kc * KC;
    // ---- stage X chunk: 128 rows x 32 cols = 1024 float4, 4 per thread ----
#pragma unroll
    for (int q = 0; q < 4; ++q) {
      const int idx = tid + 256 * q;       // 0..1023
      const int r   = idx >> 3;            // 0..127
      const int c4  = idx & 7;             // 0..7  (float4 within chunk)
      const int kk  = k0 + 4 * c4;
      const int kcl = (kk <= DDIM - 4) ? kk : (DDIM - 4);   // clamp: always in-bounds
      float4 v = *(const float4*)&X[(row0 + r) * DDIM + kcl];
      if (kk > DDIM - 4) v = make_float4(0.f, 0.f, 0.f, 0.f);  // zero K-pad lanes
      union { __bf16 h[4]; uint2 u; } pk;
      pk.h[0] = (__bf16)v.x; pk.h[1] = (__bf16)v.y;
      pk.h[2] = (__bf16)v.z; pk.h[3] = (__bf16)v.w;
      *(uint2*)&Xs[r * SPITCH + 4 * c4] = pk.u;              // 8B, aligned (72B pitch)
    }
    // ---- stage W chunk: 64 rows x 32 cols bf16, pre-padded, 8B copies ----
#pragma unroll
    for (int q = 0; q < 2; ++q) {
      const int idx = tid + 256 * q;       // 0..511
      const int n   = idx >> 3;            // 0..63
      const int c4  = idx & 7;
      uint2 w = *(const uint2*)&W1eff[n * DPAD + k0 + 4 * c4];
      *(uint2*)&Ws[n * SPITCH + 4 * c4] = w;
    }
    if (kc + 1 < DPAD / KC) {  // pull next X chunk toward L2 (global_prefetch_b8)
      __builtin_prefetch(&X[(row0 + (tid >> 1)) * DDIM + k0 + KC], 0, 1);
    }
    __syncthreads();

    // A fragment (16x32 bf16, contiguous k = 16*hl + s, as B)
    v16bf a;
    {
      const __bf16* xr = &Xs[(16 * wave + nl) * SPITCH];
#pragma unroll
      for (int j = 0; j < 8; ++j) {
        a[j]     = xr[j + 16 * hl];
        a[8 + j] = xr[8 + j + 16 * hl];
      }
    }
    // 4 N-tiles: B[k][n] = W1eff[n][k]
#pragma unroll
    for (int nt = 0; nt < 4; ++nt) {
      v16bf b;
      const __bf16* wr = &Ws[(nt * 16 + nl) * SPITCH + 16 * hl];
#pragma unroll
      for (int j = 0; j < 8; ++j) { b[2 * j] = wr[2 * j]; b[2 * j + 1] = wr[2 * j + 1]; }
      c[nt] = __builtin_amdgcn_wmma_f32_16x16x32_bf16(false, a, false, b,
                                                      (short)0, c[nt], false, false);
    }
    __syncthreads();
  }

#pragma unroll
  for (int nt = 0; nt < 4; ++nt)
#pragma unroll
    for (int r = 0; r < 8; ++r) {
      long row = row0 + 16 * wave + r + 8 * hl;
      Z[row * HDIM + nt * 16 + nl] = c[nt][r];
    }
}

// ---------------------------------------------------------------------------
// Kernel 2: serial LIF scan. 16 blocks x 4 waves; block owns 16 batch rows.
// Per step: i = z_t + y @ Wry^T (2 WMMA/wave), LIF1 on fragments,
// spikes -> LDS, readout s1 @ W2^T (waves 0-1), LIF2, accumulate t>=15.
// ---------------------------------------------------------------------------
__global__ void __launch_bounds__(128)
snn_kernel(const float* __restrict__ Z, const float* __restrict__ Wr,
           const float* __restrict__ W2, const float* __restrict__ b2,
           float* __restrict__ out) {
  __shared__ alignas(16) __bf16 ylds[16 * 66];   // 16 rows x 64 spikes, pitch 66
  const int tid  = threadIdx.x;
  const int wave = tid >> 5;
  const int lane = tid & 31;
  const int nl   = lane & 15;
  const int hl   = lane >> 4;
  const int b0   = blockIdx.x * 16;

  // Preload Wry^T B-fragments: B[k][n] = Wr[n][64+k], wave w owns N-tile w
  v16bf bw[2];
  {
    int n = wave * 16 + nl;
#pragma unroll
    for (int ch = 0; ch < 2; ++ch)
#pragma unroll
      for (int j = 0; j < 8; ++j) {
        int k = ch * 32 + 8 * hl + 2 * j + (j >= 4 ? 8 : 0);
        bw[ch][2 * j]     = (__bf16)Wr[n * 128 + 64 + k];
        bw[ch][2 * j + 1] = (__bf16)Wr[n * 128 + 64 + k + 1];
      }
  }
  // Preload W2^T B-fragments (zero-padded to N=32) and b2 (waves 0,1)
  v16bf b2f[2];
  float biasv = 0.f;
  {
    int n = wave * 16 + nl;
    bool ok = (wave < 2) && (n < NCLS);
#pragma unroll
    for (int ch = 0; ch < 2; ++ch)
#pragma unroll
      for (int j = 0; j < 8; ++j) {
        int k = ch * 32 + 8 * hl + 2 * j + (j >= 4 ? 8 : 0);
        b2f[ch][2 * j]     = (__bf16)(ok ? W2[n * HDIM + k]     : 0.f);
        b2f[ch][2 * j + 1] = (__bf16)(ok ? W2[n * HDIM + k + 1] : 0.f);
      }
    if (ok) biasv = b2[n];
  }

  v8f v1 = {}, v2 = {}, acc = {};
  for (int i = tid; i < 16 * 66; i += 128) ylds[i] = (__bf16)0.f;
  __syncthreads();

  for (int t = 0; t < TSTEPS; ++t) {
    // C-init = z_t (bias pre-folded); reads are L2-hot
    v8f c;
#pragma unroll
    for (int r = 0; r < 8; ++r) {
      int b = b0 + r + 8 * hl;
      c[r] = Z[((long)b * TSTEPS + t) * HDIM + wave * 16 + nl];
    }
    // A-fragments from previous spikes (K = 64 -> two 16x32 chunks)
    v16bf a0, a1;
    {
      const __bf16* yr = &ylds[nl * 66];
#pragma unroll
      for (int j = 0; j < 8; ++j) {
        a0[j]     = yr[j + 8 * hl];       a0[8 + j] = yr[16 + j + 8 * hl];
        a1[j]     = yr[32 + j + 8 * hl];  a1[8 + j] = yr[48 + j + 8 * hl];
      }
    }
    __syncthreads();  // all reads of prev spikes done before overwrite
    c = __builtin_amdgcn_wmma_f32_16x16x32_bf16(false, a0, false, bw[0], (short)0, c, false, false);
    c = __builtin_amdgcn_wmma_f32_16x16x32_bf16(false, a1, false, bw[1], (short)0, c, false, false);
    // LIF-1: v = v/2 + i/2; spike; hard reset. Store spikes (exact in bf16).
#pragma unroll
    for (int r = 0; r < 8; ++r) {
      float v = TAU_INV * (v1[r] + c[r]);
      float s = (v >= 1.0f) ? 1.0f : 0.0f;
      v1[r] = v * (1.0f - s);
      ylds[(r + 8 * hl) * 66 + wave * 16 + nl] = (__bf16)s;
    }
    __syncthreads();
    // Readout (wave-uniform branch: EXEC all-ones inside each executing wave)
    if (wave < 2) {
      v8f l;
#pragma unroll
      for (int r = 0; r < 8; ++r) l[r] = biasv;
      v16bf s0, s1;
      const __bf16* yr = &ylds[nl * 66];
#pragma unroll
      for (int j = 0; j < 8; ++j) {
        s0[j]     = yr[j + 8 * hl];       s0[8 + j] = yr[16 + j + 8 * hl];
        s1[j]     = yr[32 + j + 8 * hl];  s1[8 + j] = yr[48 + j + 8 * hl];
      }
      l = __builtin_amdgcn_wmma_f32_16x16x32_bf16(false, s0, false, b2f[0], (short)0, l, false, false);
      l = __builtin_amdgcn_wmma_f32_16x16x32_bf16(false, s1, false, b2f[1], (short)0, l, false, false);
#pragma unroll
      for (int r = 0; r < 8; ++r) {
        float v = TAU_INV * (v2[r] + l[r]);
        float s = (v >= 1.0f) ? 1.0f : 0.0f;
        v2[r] = v * (1.0f - s);
        if (t >= T_SKIP) acc[r] += s;
      }
    }
  }
  if (wave < 2) {
    int n = wave * 16 + nl;
    if (n < NCLS) {
#pragma unroll
      for (int r = 0; r < 8; ++r) {
        int b = b0 + r + 8 * hl;
        out[b * NCLS + n] = acc[r] * (1.0f / (float)(TSTEPS - T_SKIP));
      }
    }
  }
}

// ---------------------------------------------------------------------------
extern "C" void kernel_launch(void* const* d_in, const int* in_sizes, int n_in,
                              void* d_out, int out_size, void* d_ws, size_t ws_size,
                              hipStream_t stream) {
  const float* x  = (const float*)d_in[0];
  const float* W1 = (const float*)d_in[1];
  const float* b1 = (const float*)d_in[2];
  const float* Wr = (const float*)d_in[3];
  const float* br = (const float*)d_in[4];
  const float* W2 = (const float*)d_in[5];
  const float* b2 = (const float*)d_in[6];
  float* out = (float*)d_out;

  char* ws = (char*)d_ws;
  const size_t z_bytes = (size_t)BATCH * TSTEPS * HDIM * sizeof(float);   // 16.384 MB
  float*  Z     = (float*)ws;
  __bf16* W1eff = (__bf16*)(ws + z_bytes);
  float*  beff  = (float*)(ws + z_bytes + (size_t)HDIM * DPAD * sizeof(__bf16));

  prep_kernel<<<(HDIM * DPAD + 255) / 256, 256, 0, stream>>>(W1, b1, Wr, br, W1eff, beff);
  gemm_kernel<<<(BATCH * TSTEPS) / MBLK, 256, 0, stream>>>(x, W1eff, beff, Z);
  snn_kernel<<<BATCH / 16, 128, 0, stream>>>(Z, Wr, W2, b2, out);
}
